// RowSharNotReducRef_43258910605619
// MI455X (gfx1250) — matrix operand order repaired and checked by reading, not compiled, not run on hardware
//
#include <hip/hip_runtime.h>

// ---------------------------------------------------------------------------
// Problem constants
// ---------------------------------------------------------------------------
#define C_CLS   6
#define DF      8
#define Hh      144
#define Ww      144
#define ROW     144
#define DSi     512
#define OG      2
#define KK      5          // 2*OG+1
#define DTOK    1024
#define HEADS   16
#define DHEAD   64
#define MLP     2048
#define TIN     5760       // KK*ROW*DF
#define Bb      64
#define MH      9216       // Bb*Hh   (head GEMM M)
#define MT      384        // Bb*C_CLS (transformer M)
#define WP      148        // Ww + 2*OG
#define OUTC    292        // 2+144+2+144
#define ROWPAD  192        // cls w2 N padded to multiple of 64

// GEMM tiling
#define BM      128
#define BN      64
#define BK      32
#define PADK    40                       // LDS row pitch (elems): 20-word stride
#define A_ELEMS (BM * PADK)              // 5120
#define TILE_ELEMS ((BM + BN) * PADK)    // 7680 elems per buffer
#define BUFBYTES (TILE_ELEMS * 2)        // 15360 B
#define ABYTES   (A_ELEMS * 2)           // 10240 B

// ---------------------------------------------------------------------------
// Types / helpers
// ---------------------------------------------------------------------------
typedef __attribute__((ext_vector_type(16))) __bf16        v16bf;
typedef __attribute__((ext_vector_type(8)))  float         v8f;
typedef __attribute__((ext_vector_type(4)))  unsigned int  u32x4;

union Frag16 { v16bf v; u32x4 q[2]; };

static __device__ __forceinline__ unsigned short f32_to_bf16(float f) {
  unsigned int u = __float_as_uint(f);
  u += 0x7FFFu + ((u >> 16) & 1u);        // round-to-nearest-even
  return (unsigned short)(u >> 16);
}
static __device__ __forceinline__ float bf16_to_f32(unsigned short s) {
  return __uint_as_float(((unsigned int)s) << 16);
}

// CDNA5 async global->LDS copy (16B per lane), tracked by ASYNCcnt.
static __device__ __forceinline__ void async_b128(unsigned lds_addr,
                                                  const unsigned short* gsrc) {
  asm volatile("global_load_async_to_lds_b128 %0, %1, off"
               :: "v"(lds_addr), "v"(gsrc) : "memory");
}
static __device__ __forceinline__ void wait_async0() {
  asm volatile("s_wait_asynccnt 0" ::: "memory");
}

// ---------------------------------------------------------------------------
// LDS fragment gathers (layouts per CDNA5 ISA 7.12.2)
// A 16x32 bf16: lane L row m=L%16; chunks k=(L<16?0:8)+{0..7} and +16
// B 32x16 bf16: lane L col n=L%16; chunks k=(L<16?0:8)+{0..7} and +16
// ---------------------------------------------------------------------------
static __device__ __forceinline__ void lds_afrag(Frag16& f,
    const unsigned short* base, int mBase, int lane) {
  const int ml = lane & 15, half = lane >> 4;
  const unsigned short* p = base + (mBase + ml) * PADK + half * 8;
  f.q[0] = *(const u32x4*)(p);
  f.q[1] = *(const u32x4*)(p + 16);
}
static __device__ __forceinline__ void lds_bfrag(Frag16& f,
    const unsigned short* base, int nBase, int lane) {
  const int nl = lane & 15, kh = lane >> 4;
  const unsigned short* p = base + (nBase + nl) * PADK + kh * 8;
  f.q[0] = *(const u32x4*)(p);
  f.q[1] = *(const u32x4*)(p + 16);
}

// ---------------------------------------------------------------------------
// bf16 WMMA GEMM with async-LDS double buffering:
//   out[m][n] = epilogue( sum_k A[m][k] * Bt[n][k] )
//   A: M x K row-major bf16; Bt: Npad x K row-major bf16 (rows >= N zero)
//   grid = (Npad/64, M/128, nClass); block 256 = 8 waves, wave -> 32x32 tile
//   modes: 0 outF=acc+bias | 1 outB=bf16((acc+bias)*g'+be) | 3 outB=bf16(gelu)
//          4 outF=acc+bias+res | 5 outF=acc+bias+emb[(m%6)*N+n]
// ---------------------------------------------------------------------------
__global__ void __launch_bounds__(256)
gemm_bf16_kernel(const unsigned short* __restrict__ A,
                 const unsigned short* __restrict__ Bt,
                 int N, int Kd,
                 long sA, long sB, long sOut, long sBias,
                 const float* __restrict__ bias,
                 const float* __restrict__ g,
                 const float* __restrict__ be,
                 const float* __restrict__ res,
                 const float* __restrict__ emb,
                 float* __restrict__ outF,
                 unsigned short* __restrict__ outB,
                 int mode) {
  __shared__ unsigned short tiles[2][TILE_ELEMS];

  const int tid  = threadIdx.x;
  const int lane = tid & 31;
  const int wave = tid >> 5;
  const int wm = wave & 3;        // 4 m-groups of 32 rows
  const int wn = wave >> 2;       // 2 n-groups of 32 cols
  const int z = blockIdx.z;

  const unsigned short* gA = A + (long)z * sA + (long)blockIdx.y * BM * Kd;
  const unsigned short* gB = Bt + (long)z * sB + (long)blockIdx.x * BN * Kd;
  if (bias) bias += (long)z * sBias;
  const float* gz  = g  ? g  + (long)z * sBias : nullptr;
  const float* bez = be ? be + (long)z * sBias : nullptr;
  const long outOff = (long)z * sOut;

  // staging assignment: 3 x 16B chunks per thread per k-step
  const int trow = tid >> 2;            // 0..63
  const int tcol = (tid & 3) * 8;       // 0,8,16,24
  const unsigned ldsbase = (unsigned)(size_t)&tiles[0][0];

  const int nk = Kd / BK;
#define STAGE(buf, k0)                                                         \
  do {                                                                         \
    const unsigned bb = ldsbase + (unsigned)(buf) * BUFBYTES;                  \
    async_b128(bb + (unsigned)((trow * PADK + tcol) * 2),                      \
               gA + (long)trow * Kd + (k0) + tcol);                            \
    async_b128(bb + (unsigned)(((trow + 64) * PADK + tcol) * 2),               \
               gA + (long)(trow + 64) * Kd + (k0) + tcol);                     \
    async_b128(bb + (unsigned)(ABYTES + (trow * PADK + tcol) * 2),             \
               gB + (long)trow * Kd + (k0) + tcol);                            \
  } while (0)

  STAGE(0, 0);

  v8f acc0 = {0.f,0.f,0.f,0.f,0.f,0.f,0.f,0.f};
  v8f acc1 = {0.f,0.f,0.f,0.f,0.f,0.f,0.f,0.f};
  v8f acc2 = {0.f,0.f,0.f,0.f,0.f,0.f,0.f,0.f};
  v8f acc3 = {0.f,0.f,0.f,0.f,0.f,0.f,0.f,0.f};

  for (int kt = 0; kt < nk; ++kt) {
    wait_async0();
    __syncthreads();
    if (kt + 1 < nk) STAGE((kt + 1) & 1, (kt + 1) * BK);

    const unsigned short* At = &tiles[kt & 1][0];
    const unsigned short* Bl = At + A_ELEMS;
    Frag16 a0, a1, b0, b1;
    lds_afrag(a0, At, wm * 32,      lane);
    lds_afrag(a1, At, wm * 32 + 16, lane);
    lds_bfrag(b0, Bl, wn * 32,      lane);
    lds_bfrag(b1, Bl, wn * 32 + 16, lane);
    acc0 = __builtin_amdgcn_wmma_f32_16x16x32_bf16(false, a0.v, false, b0.v,
                                                   (short)0, acc0, false, false);
    acc1 = __builtin_amdgcn_wmma_f32_16x16x32_bf16(false, a0.v, false, b1.v,
                                                   (short)0, acc1, false, false);
    acc2 = __builtin_amdgcn_wmma_f32_16x16x32_bf16(false, a1.v, false, b0.v,
                                                   (short)0, acc2, false, false);
    acc3 = __builtin_amdgcn_wmma_f32_16x16x32_bf16(false, a1.v, false, b1.v,
                                                   (short)0, acc3, false, false);
  }
#undef STAGE

  // epilogue (C layout: elem i <-> m = i + 8*(lane>=16); n = lane%16)
  const int nl = lane & 15, half = lane >> 4;
  const float RS = 0.9999950000374997f;   // 1/sqrt(1+1e-5)  (BN fold)
#pragma unroll
  for (int fi = 0; fi < 4; ++fi) {
    const int mi = fi >> 1, ni = fi & 1;
    v8f cc = (fi == 0) ? acc0 : (fi == 1) ? acc1 : (fi == 2) ? acc2 : acc3;
    const int n = blockIdx.x * BN + wn * 32 + ni * 16 + nl;
    if (n >= N) continue;
    const int mB = blockIdx.y * BM + wm * 32 + mi * 16 + half * 8;
#pragma unroll
    for (int i = 0; i < 8; ++i) {
      const int m = mB + i;
      float val = cc[i] + (bias ? bias[n] : 0.f);
      const long oi = outOff + (long)m * N + n;
      if (mode == 1) {
        val = val * (gz[n] * RS) + bez[n];
        outB[oi] = f32_to_bf16(val);
      } else if (mode == 3) {
        val = 0.5f * val * (1.f + erff(val * 0.7071067811865475f));
        outB[oi] = f32_to_bf16(val);
      } else {
        if (mode == 4)      val += res[(long)m * N + n];
        else if (mode == 5) val += emb[(long)(m % C_CLS) * N + n];
        outF[oi] = val;
      }
    }
  }
}

// ---------------------------------------------------------------------------
// Elementwise / packing kernels
// ---------------------------------------------------------------------------
__global__ void f2bf_kernel(const float* __restrict__ src,
                            unsigned short* __restrict__ dst, long n) {
  long i = (long)blockIdx.x * blockDim.x + threadIdx.x;
  const long stride = (long)gridDim.x * blockDim.x;
  for (; i < n; i += stride) dst[i] = f32_to_bf16(src[i]);
}

// cls w2 (C,144,512) -> padded bf16 (C,192,512), rows 144..191 zero
__global__ void pack_w2pad_kernel(const float* __restrict__ src,
                                  unsigned short* __restrict__ dst, long n) {
  long i = (long)blockIdx.x * blockDim.x + threadIdx.x;
  const long stride = (long)gridDim.x * blockDim.x;
  for (; i < n; i += stride) {
    const int o  = (int)(i % DSi);
    const int np = (int)((i / DSi) % ROWPAD);
    const int c  = (int)(i / ((long)DSi * ROWPAD));
    dst[i] = (np < ROW) ? f32_to_bf16(src[((long)c * ROW + np) * DSi + o])
                        : (unsigned short)0;
  }
}

// A1[m=b*H+h][d=f*W+w] = bf16(x[b,f,h,w])
__global__ void pack_x_kernel(const float* __restrict__ x,
                              unsigned short* __restrict__ a1, long n) {
  long i = (long)blockIdx.x * blockDim.x + threadIdx.x;
  const long stride = (long)gridDim.x * blockDim.x;
  for (; i < n; i += stride) {
    const int d = (int)(i % 1152);
    const long m = i / 1152;
    const int f = d / Ww, w = d % Ww;
    const int b = (int)(m / Hh), h = (int)(m % Hh);
    a1[i] = f32_to_bf16(x[(((long)b * DF + f) * Hh + h) * Ww + w]);
  }
}

// padT[b][h][wp][f]  ( WP = W + 4 ), zero padding on wp
__global__ void build_padT_kernel(const float* __restrict__ x,
                                  float* __restrict__ padT, long n) {
  long i = (long)blockIdx.x * blockDim.x + threadIdx.x;
  const long stride = (long)gridDim.x * blockDim.x;
  for (; i < n; i += stride) {
    const int f  = (int)(i % DF);
    const int wp = (int)((i / DF) % WP);
    const int h  = (int)((i / ((long)DF * WP)) % Hh);
    const int b  = (int)(i / ((long)DF * WP * Hh));
    float v = 0.f;
    if (wp >= OG && wp < Ww + OG)
      v = x[(((long)b * DF + f) * Hh + h) * Ww + (wp - OG)];
    padT[i] = v;
  }
}

// A1[m][d=f*W+w] = bf16(padT[b][h][w+OG][f])   (second-pass rt2)
__global__ void pack_padT_kernel(const float* __restrict__ padT,
                                 unsigned short* __restrict__ a1, long n) {
  long i = (long)blockIdx.x * blockDim.x + threadIdx.x;
  const long stride = (long)gridDim.x * blockDim.x;
  for (; i < n; i += stride) {
    const int d = (int)(i % 1152);
    const long m = i / 1152;
    const int f = d / Ww, w = d % Ww;
    a1[i] = f32_to_bf16(padT[((long)m * WP + (w + OG)) * DF + f]);
  }
}

// ext head: z = Ybf @ w2^T (N=2) + b2, softmax2 -> out cols, p0 buffer
__global__ void ext_head_kernel(const unsigned short* __restrict__ Ybf,
                                const float* __restrict__ w2,
                                const float* __restrict__ b2,
                                float* __restrict__ out,
                                float* __restrict__ p0buf, int colOff) {
  const long idx = (long)blockIdx.x * blockDim.x + threadIdx.x;  // c*MH+m
  if (idx >= (long)C_CLS * MH) return;
  const int c = (int)(idx / MH), m = (int)(idx % MH);
  const unsigned short* y = Ybf + idx * DSi;
  const float* w0 = w2 + (long)c * 2 * DSi;
  const float* w1 = w0 + DSi;
  float a0 = 0.f, a1 = 0.f;
  for (int o = 0; o < DSi; ++o) {
    const float yv = bf16_to_f32(y[o]);
    a0 += yv * w0[o];
    a1 += yv * w1[o];
  }
  a0 += b2[c * 2 + 0];
  a1 += b2[c * 2 + 1];
  const float mx = fmaxf(a0, a1);
  const float e0 = expf(a0 - mx), e1 = expf(a1 - mx);
  const float p0 = e0 / (e0 + e1);
  const int b = m / Hh, h = m % Hh;
  const long ob = (((long)(c * Bb + b)) * Hh + h) * OUTC + colOff;
  out[ob] = p0;
  out[ob + 1] = e1 / (e0 + e1);
  p0buf[idx] = p0;
}

// kmask[b*6+c] = mean_h(p0[c,b,h]) > 0.3
__global__ void sel_kernel(const float* __restrict__ p0,
                           int* __restrict__ kmask) {
  const int idx = blockIdx.x * blockDim.x + threadIdx.x;  // c*Bb + b
  if (idx >= C_CLS * Bb) return;
  const int c = idx / Bb, b = idx % Bb;
  float s = 0.f;
  for (int h = 0; h < Hh; ++h) s += p0[(long)c * MH + b * Hh + h];
  kmask[b * C_CLS + c] = (s * (1.f / Hh) > 0.3f) ? 1 : 0;
}

// softmax over 144 + argmax;  r = c*MH + m, block of 256 threads
__global__ void softmax_cls_kernel(const float* __restrict__ z,
                                   float* __restrict__ out,
                                   int* __restrict__ corr, int colOff) {
  const int r = blockIdx.x;
  const int t = threadIdx.x;
  __shared__ float sm[256];
  __shared__ int   si[256];
  const float* zr = z + (long)r * ROW;
  const float v = (t < ROW) ? zr[t] : -3.4e38f;
  sm[t] = v; si[t] = (t < ROW) ? t : 0;
  __syncthreads();
  for (int o = 128; o > 0; o >>= 1) {
    if (t < o && sm[t + o] > sm[t]) { sm[t] = sm[t + o]; si[t] = si[t + o]; }
    __syncthreads();
  }
  const float mx = sm[0];
  const int amax = si[0];
  __syncthreads();
  const float e = (t < ROW) ? expf(v - mx) : 0.f;
  sm[t] = e;
  __syncthreads();
  for (int o = 128; o > 0; o >>= 1) {
    if (t < o) sm[t] += sm[t + o];
    __syncthreads();
  }
  const float inv = 1.f / sm[0];
  const int c = r / MH, m = r % MH, b = m / Hh, h = m % Hh;
  if (t < ROW)
    out[(((long)(c * Bb + b)) * Hh + h) * OUTC + colOff + t] = e * inv;
  if (t == 0 && corr) corr[r] = amax;
}

// tok_in[b*6+c][(f*H+h)*K+k] = bf16(padT[b][h][corr+k][f])
__global__ void gather_tok_kernel(const float* __restrict__ padT,
                                  const int* __restrict__ corr,
                                  unsigned short* __restrict__ tokin, long n) {
  long i = (long)blockIdx.x * blockDim.x + threadIdx.x;
  const long stride = (long)gridDim.x * blockDim.x;
  for (; i < n; i += stride) {
    const int j = (int)(i % TIN);
    const int m = (int)(i / TIN);
    const int b = m / C_CLS, c = m % C_CLS;
    const int k = j % KK;
    const int fh = j / KK;
    const int f = fh / Hh, h = fh % Hh;
    const int wpx = corr[(long)c * MH + b * Hh + h] + k;
    tokin[i] = f32_to_bf16(padT[((long)(b * Hh + h) * WP + wpx) * DF + f]);
  }
}

// LayerNorm over `width`, fp32 in -> bf16 out.  grid = rows, block = 256
__global__ void ln_kernel(const float* __restrict__ in,
                          const float* __restrict__ gsc,
                          const float* __restrict__ bsc,
                          unsigned short* __restrict__ outbf, int width) {
  const int row = blockIdx.x, t = threadIdx.x;
  __shared__ float red[256];
  const float* r = in + (long)row * width;
  float s = 0.f;
  for (int i = t; i < width; i += 256) s += r[i];
  red[t] = s; __syncthreads();
  for (int o = 128; o > 0; o >>= 1) { if (t < o) red[t] += red[t + o]; __syncthreads(); }
  const float mean = red[0] / width;
  __syncthreads();
  float vs = 0.f;
  for (int i = t; i < width; i += 256) { const float d = r[i] - mean; vs += d * d; }
  red[t] = vs; __syncthreads();
  for (int o = 128; o > 0; o >>= 1) { if (t < o) red[t] += red[t + o]; __syncthreads(); }
  const float inv = rsqrtf(red[0] / width + 1e-5f);
  for (int i = t; i < width; i += 256)
    outbf[(long)row * width + i] = f32_to_bf16((r[i] - mean) * inv * gsc[i] + bsc[i]);
}

// 6-token masked attention.  grid=(Bb, HEADS), block=64
__global__ void attn_kernel(const float* __restrict__ qkv,
                            const int* __restrict__ kmask,
                            unsigned short* __restrict__ ao_bf) {
  const int b = blockIdx.x, hh = blockIdx.y, t = threadIdx.x;
  __shared__ float q[C_CLS][DHEAD], k[C_CLS][DHEAD], v[C_CLS][DHEAD];
  __shared__ float att[C_CLS][C_CLS];
  for (int n = 0; n < C_CLS; ++n) {
    const long base = ((long)(b * C_CLS + n)) * (3 * DTOK) + hh * DHEAD + t;
    q[n][t] = qkv[base];
    k[n][t] = qkv[base + DTOK];
    v[n][t] = qkv[base + 2 * DTOK];
  }
  __syncthreads();
  if (t < C_CLS * C_CLS) {
    const int n = t / C_CLS, m = t % C_CLS;
    float s = 0.f;
    for (int d = 0; d < DHEAD; ++d) s += q[n][d] * k[m][d];
    s *= 0.125f;                                   // 1/sqrt(64)
    if (!kmask[b * C_CLS + m]) s = -1e9f;
    att[n][m] = s;
  }
  __syncthreads();
  if (t < C_CLS) {
    float mx = -3.4e38f;
    for (int m = 0; m < C_CLS; ++m) mx = fmaxf(mx, att[t][m]);
    float e[C_CLS], sum = 0.f;
    for (int m = 0; m < C_CLS; ++m) { e[m] = expf(att[t][m] - mx); sum += e[m]; }
    for (int m = 0; m < C_CLS; ++m) att[t][m] = e[m] / sum;
  }
  __syncthreads();
  for (int n = 0; n < C_CLS; ++n) {
    float s = 0.f;
    for (int m = 0; m < C_CLS; ++m) s += att[n][m] * v[m][t];
    ao_bf[((long)(b * C_CLS + n)) * DTOK + hh * DHEAD + t] = f32_to_bf16(s);
  }
}

// scatter out5 back into padT, serialized over classes.  grid=Bb*Hh, block=64
__global__ void scatter_kernel(const float* __restrict__ out5,
                               const int* __restrict__ corr,
                               const int* __restrict__ kmask,
                               float* __restrict__ padT) {
  const int bh = blockIdx.x;
  const int b = bh / Hh, h = bh % Hh;
  const int t = threadIdx.x;
  for (int c = 0; c < C_CLS; ++c) {
    const bool cond = (kmask[b * C_CLS + c] != 0) && (h < Hh - 1);
    if (cond && t < KK * DF) {
      const int k = t / DF, f = t % DF;
      const int wpx = corr[(long)c * MH + bh] + k;
      padT[((long)bh * WP + wpx) * DF + f] =
          out5[((long)(b * C_CLS + c)) * TIN + (f * ROW + h) * KK + k];
    }
    __syncthreads();
  }
}

// ---------------------------------------------------------------------------
// Host orchestration
// ---------------------------------------------------------------------------
static void launch_gemm(hipStream_t s, const unsigned short* A,
                        const unsigned short* Bt, int M, int N, int Npad, int Kd,
                        long sA, long sB, long sOut, long sBias,
                        const float* bias, const float* g, const float* be,
                        const float* res, const float* emb,
                        float* outF, unsigned short* outB, int mode, int nc) {
  dim3 grid(Npad / BN, M / BM, nc);
  gemm_bf16_kernel<<<grid, 256, 0, s>>>(A, Bt, N, Kd, sA, sB, sOut, sBias,
                                        bias, g, be, res, emb, outF, outB, mode);
}
static void launch_f2bf(hipStream_t s, const float* src, unsigned short* dst, long n) {
  int blocks = (int)((n + 255) / 256); if (blocks > 4096) blocks = 4096;
  f2bf_kernel<<<blocks, 256, 0, s>>>(src, dst, n);
}

extern "C" void kernel_launch(void* const* d_in, const int* in_sizes, int n_in,
                              void* d_out, int out_size, void* d_ws, size_t ws_size,
                              hipStream_t stream) {
  (void)in_sizes; (void)n_in; (void)out_size; (void)ws_size;
  const float* x = (const float*)d_in[0];
  const float* ext_p[6];  for (int i = 0; i < 6; ++i) ext_p[i]  = (const float*)d_in[1 + i];
  const float* cls_p[6];  for (int i = 0; i < 6; ++i) cls_p[i]  = (const float*)d_in[7 + i];
  const float* ext2_p[6]; for (int i = 0; i < 6; ++i) ext2_p[i] = (const float*)d_in[13 + i];
  const float* cls2_p[6]; for (int i = 0; i < 6; ++i) cls2_p[i] = (const float*)d_in[19 + i];
  const float* tok_w   = (const float*)d_in[25];
  const float* tok_b   = (const float*)d_in[26];
  const float* emb     = (const float*)d_in[27];
  const float* ln1_g   = (const float*)d_in[28];
  const float* ln1_b   = (const float*)d_in[29];
  const float* qkv_w   = (const float*)d_in[30];
  const float* attno_w = (const float*)d_in[31];
  const float* attno_b = (const float*)d_in[32];
  const float* ln2_g   = (const float*)d_in[33];
  const float* ln2_b   = (const float*)d_in[34];
  const float* ff_w1   = (const float*)d_in[35];
  const float* ff_b1   = (const float*)d_in[36];
  const float* ff_w2   = (const float*)d_in[37];
  const float* ff_b2   = (const float*)d_in[38];
  const float* lnf_g   = (const float*)d_in[39];
  const float* lnf_b   = (const float*)d_in[40];
  const float* proj_w  = (const float*)d_in[41];
  const float* proj_b  = (const float*)d_in[42];
  float* out = (float*)d_out;

  // ---- workspace layout ----
  char* wsp = (char*)d_ws;
  auto alloc = [&](size_t bytes) -> char* {
    char* r = wsp; wsp += (bytes + 255) & ~(size_t)255; return r;
  };
  unsigned short* a1bf   = (unsigned short*)alloc((size_t)MH * 1152 * 2);
  unsigned short* w1e    = (unsigned short*)alloc((size_t)C_CLS * DSi * 1152 * 2);
  unsigned short* w1c    = (unsigned short*)alloc((size_t)C_CLS * DSi * 1152 * 2);
  unsigned short* w1e2   = (unsigned short*)alloc((size_t)C_CLS * DSi * 1152 * 2);
  unsigned short* w1c2   = (unsigned short*)alloc((size_t)C_CLS * DSi * 1152 * 2);
  unsigned short* w2c    = (unsigned short*)alloc((size_t)C_CLS * ROWPAD * DSi * 2);
  unsigned short* w2c2   = (unsigned short*)alloc((size_t)C_CLS * ROWPAD * DSi * 2);
  unsigned short* ybf    = (unsigned short*)alloc((size_t)C_CLS * MH * DSi * 2);
  float*          zcls   = (float*)alloc((size_t)C_CLS * MH * ROW * 4);
  float*          p0buf  = (float*)alloc((size_t)C_CLS * MH * 4);
  int*            kmask  = (int*)alloc((size_t)Bb * C_CLS * 4);
  int*            corr   = (int*)alloc((size_t)C_CLS * MH * 4);
  float*          padT   = (float*)alloc((size_t)Bb * Hh * WP * DF * 4);
  unsigned short* tokin  = (unsigned short*)alloc((size_t)MT * TIN * 2);
  unsigned short* tokwbf = (unsigned short*)alloc((size_t)DTOK * TIN * 2);
  unsigned short* qkvwbf = (unsigned short*)alloc((size_t)3 * DTOK * DTOK * 2);
  unsigned short* aowbf  = (unsigned short*)alloc((size_t)DTOK * DTOK * 2);
  unsigned short* f1wbf  = (unsigned short*)alloc((size_t)MLP * DTOK * 2);
  unsigned short* f2wbf  = (unsigned short*)alloc((size_t)DTOK * MLP * 2);
  unsigned short* pjwbf  = (unsigned short*)alloc((size_t)TIN * DTOK * 2);
  float*          seq    = (float*)alloc((size_t)MT * DTOK * 4);
  unsigned short* hnbf   = (unsigned short*)alloc((size_t)MT * DTOK * 2);
  float*          qkvo   = (float*)alloc((size_t)MT * 3 * DTOK * 4);
  unsigned short* aobf   = (unsigned short*)alloc((size_t)MT * DTOK * 2);
  unsigned short* ffbf   = (unsigned short*)alloc((size_t)MT * MLP * 2);
  unsigned short* sfbf   = (unsigned short*)alloc((size_t)MT * DTOK * 2);
  float*          out5   = (float*)alloc((size_t)MT * TIN * 4);

  // ---- pack inputs / weights to bf16 ----
  {
    pack_x_kernel<<<4096, 256, 0, stream>>>(x, a1bf, (long)MH * 1152);
    const long nW1 = (long)C_CLS * DSi * 1152;
    launch_f2bf(stream, ext_p[0],  w1e,  nW1);
    launch_f2bf(stream, cls_p[0],  w1c,  nW1);
    launch_f2bf(stream, ext2_p[0], w1e2, nW1);
    launch_f2bf(stream, cls2_p[0], w1c2, nW1);
    const long nW2 = (long)C_CLS * ROWPAD * DSi;
    pack_w2pad_kernel<<<(int)((nW2 + 255) / 256), 256, 0, stream>>>(cls_p[4],  w2c,  nW2);
    pack_w2pad_kernel<<<(int)((nW2 + 255) / 256), 256, 0, stream>>>(cls2_p[4], w2c2, nW2);
    launch_f2bf(stream, tok_w,   tokwbf, (long)DTOK * TIN);
    launch_f2bf(stream, qkv_w,   qkvwbf, (long)3 * DTOK * DTOK);
    launch_f2bf(stream, attno_w, aowbf,  (long)DTOK * DTOK);
    launch_f2bf(stream, ff_w1,   f1wbf,  (long)MLP * DTOK);
    launch_f2bf(stream, ff_w2,   f2wbf,  (long)DTOK * MLP);
    launch_f2bf(stream, proj_w,  pjwbf,  (long)TIN * DTOK);
    build_padT_kernel<<<4096, 256, 0, stream>>>(x, padT, (long)Bb * Hh * WP * DF);
  }

  const long sW1 = (long)DSi * 1152, sY = (long)MH * DSi;
  const long sW2 = (long)ROWPAD * DSi, sZ = (long)MH * ROW;

  // ---- head pass 1: ext ----
  launch_gemm(stream, a1bf, w1e, MH, DSi, DSi, 1152, 0, sW1, sY, DSi,
              ext_p[1], ext_p[2], ext_p[3], nullptr, nullptr,
              nullptr, ybf, 1, C_CLS);
  ext_head_kernel<<<(C_CLS * MH + 255) / 256, 256, 0, stream>>>(
      ybf, ext_p[4], ext_p[5], out, p0buf, 0);
  sel_kernel<<<2, 256, 0, stream>>>(p0buf, kmask);

  // ---- head pass 1: cls ----
  launch_gemm(stream, a1bf, w1c, MH, DSi, DSi, 1152, 0, sW1, sY, DSi,
              cls_p[1], cls_p[2], cls_p[3], nullptr, nullptr,
              nullptr, ybf, 1, C_CLS);
  launch_gemm(stream, ybf, w2c, MH, ROW, ROWPAD, DSi, sY, sW2, sZ, ROW,
              cls_p[5], nullptr, nullptr, nullptr, nullptr,
              zcls, nullptr, 0, C_CLS);
  softmax_cls_kernel<<<C_CLS * MH, 256, 0, stream>>>(zcls, out, corr, 2);

  // ---- token gather + projection ----
  gather_tok_kernel<<<4096, 256, 0, stream>>>(padT, corr, tokin, (long)MT * TIN);
  launch_gemm(stream, tokin, tokwbf, MT, DTOK, DTOK, TIN, 0, 0, 0, 0,
              tok_b, nullptr, nullptr, nullptr, emb, seq, nullptr, 5, 1);

  // ---- transformer ----
  ln_kernel<<<MT, 256, 0, stream>>>(seq, ln1_g, ln1_b, hnbf, DTOK);
  launch_gemm(stream, hnbf, qkvwbf, MT, 3 * DTOK, 3 * DTOK, DTOK, 0, 0, 0, 0,
              nullptr, nullptr, nullptr, nullptr, nullptr, qkvo, nullptr, 0, 1);
  attn_kernel<<<dim3(Bb, HEADS), DHEAD, 0, stream>>>(qkvo, kmask, aobf);
  launch_gemm(stream, aobf, aowbf, MT, DTOK, DTOK, DTOK, 0, 0, 0, 0,
              attno_b, nullptr, nullptr, seq, nullptr, seq, nullptr, 4, 1);
  ln_kernel<<<MT, 256, 0, stream>>>(seq, ln2_g, ln2_b, hnbf, DTOK);
  launch_gemm(stream, hnbf, f1wbf, MT, MLP, MLP, DTOK, 0, 0, 0, 0,
              ff_b1, nullptr, nullptr, nullptr, nullptr, nullptr, ffbf, 3, 1);
  launch_gemm(stream, ffbf, f2wbf, MT, DTOK, DTOK, MLP, 0, 0, 0, 0,
              ff_b2, nullptr, nullptr, seq, nullptr, seq, nullptr, 4, 1);
  ln_kernel<<<MT, 256, 0, stream>>>(seq, lnf_g, lnf_b, sfbf, DTOK);
  launch_gemm(stream, sfbf, pjwbf, MT, TIN, TIN, DTOK, 0, 0, 0, 0,
              proj_b, nullptr, nullptr, nullptr, nullptr, out5, nullptr, 0, 1);

  // ---- scatter back into padT (serialized over classes) ----
  scatter_kernel<<<Bb * Hh, 64, 0, stream>>>(out5, corr, kmask, padT);

  // ---- head pass 2 on updated tensor ----
  pack_padT_kernel<<<4096, 256, 0, stream>>>(padT, a1bf, (long)MH * 1152);
  launch_gemm(stream, a1bf, w1e2, MH, DSi, DSi, 1152, 0, sW1, sY, DSi,
              ext2_p[1], ext2_p[2], ext2_p[3], nullptr, nullptr,
              nullptr, ybf, 1, C_CLS);
  ext_head_kernel<<<(C_CLS * MH + 255) / 256, 256, 0, stream>>>(
      ybf, ext2_p[4], ext2_p[5], out, p0buf, 146);
  launch_gemm(stream, a1bf, w1c2, MH, DSi, DSi, 1152, 0, sW1, sY, DSi,
              cls2_p[1], cls2_p[2], cls2_p[3], nullptr, nullptr,
              nullptr, ybf, 1, C_CLS);
  launch_gemm(stream, ybf, w2c2, MH, ROW, ROWPAD, DSi, sY, sW2, sZ, ROW,
              cls2_p[5], nullptr, nullptr, nullptr, nullptr,
              zcls, nullptr, 0, C_CLS);
  softmax_cls_kernel<<<C_CLS * MH, 256, 0, stream>>>(zcls, out, nullptr, 148);
}
